// GraphTransformerEncoder_68667937129205
// MI455X (gfx1250) — compile-verified, matrix operand order checked by reading, not run on hardware
//
#include <hip/hip_runtime.h>
#include <cstdint>

typedef __attribute__((ext_vector_type(16))) _Float16 v16h;
typedef __attribute__((ext_vector_type(8)))  float    v8f;

#define B_GRAPHS 256
#define N_NODES  128
#define HID      256
#define HEADS    8
#define DH       32
#define E_PER_G  2048
#define E_TOT    (B_GRAPHS * E_PER_G)
#define NN       (B_GRAPHS * N_NODES)
#define K_RW     8

// ---------------------------------------------------------------------------
// WMMA fragment helper (CDNA5 wave32, V_WMMA_F32_16X16X32_F16).
// 16-bit A/B fragment per-lane layout (ISA 7.12.2):
//   lane: r = lane&15 (row of A / col of B), half = lane>>4
//   element j: k = half*8 + j        for j in [0,8)
//              k = 16 + half*8 + j-8 for j in [8,16)
// Both 8-element runs are contiguous in K -> two 16B (b128) loads.
// ---------------------------------------------------------------------------
struct alignas(16) bytes16 { uint32_t w[4]; };

__device__ inline v16h frag_load_contig(const _Float16* row, int half) {
  union { v16h v; bytes16 q[2]; } u;
  u.q[0] = *(const bytes16*)(row + half * 8);
  u.q[1] = *(const bytes16*)(row + 16 + half * 8);
  return u.v;
}

__device__ inline v8f wmma_f16(v16h a, v16h b, v8f c) {
  return __builtin_amdgcn_wmma_f32_16x16x32_f16(false, a, false, b, (short)0, c,
                                                false, false);
}

// ---------------------------------------------------------------------------
// Kernel 1: per-graph random-walk PE, fully LDS-resident.
// Adjacency symmetry: trans[k][n] = adj[n][k]*dinv[n], so row-scaling the
// symmetric adjacency gives the Bt layout directly (contiguous K reads).
// Wave owns a 16x128 strip; A fragment hoisted per k-step; B fragments
// software-pipelined (distance 1) so LDS loads overlap the XDL pipe.
// LDS: 3 * 128*136 f16 (~102 KB) of the 320 KB WGP pool.
// ---------------------------------------------------------------------------
#define LP 136  // padded row stride (272 B -> 16B-aligned, bank-friendly)

__global__ void __launch_bounds__(256) rw_pe_kernel(
    const int* __restrict__ src, const int* __restrict__ dst,
    int* __restrict__ degree_out, float* __restrict__ rw_pe) {
  __shared__ __align__(16) _Float16 T [N_NODES * LP];  // adj -> row-scaled adj
  __shared__ __align__(16) _Float16 P0[N_NODES * LP];
  __shared__ __align__(16) _Float16 P1[N_NODES * LP];
  __shared__ int   degi[N_NODES];
  __shared__ float dinv[N_NODES];

  const int b = blockIdx.x;
  const int tid = threadIdx.x;

  if (tid == 0) {
    __builtin_prefetch(src + b * E_PER_G, 0, 1);
    __builtin_prefetch(dst + b * E_PER_G, 0, 1);
  }

  for (int i = tid; i < N_NODES * LP; i += 256) T[i] = (_Float16)0.f;
  if (tid < N_NODES) degi[tid] = 0;
  __syncthreads();

  for (int e = tid; e < E_PER_G; e += 256) {
    const int ge = b * E_PER_G + e;
    const int s = src[ge] - b * N_NODES;
    const int d = dst[ge] - b * N_NODES;
    T[s * LP + d] = (_Float16)1.f;  // symmetrize: (adj+adjT)>0
    T[d * LP + s] = (_Float16)1.f;
    atomicAdd(&degi[d], 1);         // directed in-degree for deg_emb
  }
  __syncthreads();

  if (tid < N_NODES) {
    float srow = 0.f;
    for (int j = 0; j < N_NODES; ++j) srow += (float)T[tid * LP + j];
    dinv[tid] = srow > 0.f ? 1.f / srow : 0.f;
  }
  __syncthreads();

  // Row-scale: T[n][k] = adj[n][k]*dinv[n] == trans[k][n]  (Bt layout)
  for (int i = tid; i < N_NODES * N_NODES; i += 256) {
    const int row = i >> 7, col = i & 127;
    T[row * LP + col]  = (_Float16)((float)T[row * LP + col] * dinv[row]);
    P0[row * LP + col] = (_Float16)(row == col ? 1.f : 0.f);
  }
  __syncthreads();

  _Float16* Pin = P0;
  _Float16* Pout = P1;
  const int wave = tid >> 5, lane = tid & 31, half = lane >> 4, r = lane & 15;

  for (int step = 0; step < K_RW; ++step) {
    v8f acc[8] = {};
    const _Float16* arow = Pin + (wave * 16 + r) * LP;
#pragma unroll
    for (int k0 = 0; k0 < N_NODES; k0 += 32) {
      v16h a = frag_load_contig(arow + k0, half);
      // B software pipeline: issue load t+1 before WMMA t
      v16h bb = frag_load_contig(T + r * LP + k0, half);
#pragma unroll
      for (int t = 0; t < 7; ++t) {
        v16h bn = frag_load_contig(T + ((t + 1) * 16 + r) * LP + k0, half);
        acc[t] = wmma_f16(a, bb, acc[t]);
        bb = bn;
      }
      acc[7] = wmma_f16(a, bb, acc[7]);
    }
#pragma unroll
    for (int t = 0; t < 8; ++t)
#pragma unroll
      for (int e2 = 0; e2 < 8; ++e2)
        Pout[(wave * 16 + e2 + 8 * half) * LP + t * 16 + r] = (_Float16)acc[t][e2];
    __syncthreads();
    if (tid < N_NODES)
      rw_pe[(b * N_NODES + tid) * K_RW + step] = (float)Pout[tid * LP + tid];
    _Float16* tmp = Pin; Pin = Pout; Pout = tmp;
    __syncthreads();
  }
  if (tid < N_NODES) degree_out[b * N_NODES + tid] = degi[tid];
}

// ---------------------------------------------------------------------------
// Kernel 2: transpose+convert a 256x256 f32 weight into f16 Wt[n][k]=W[k][n]
// so B-fragments load contiguously along K.
// ---------------------------------------------------------------------------
__global__ void wt_f16_kernel(const float* __restrict__ W, _Float16* __restrict__ Wt) {
  const int idx = blockIdx.x * blockDim.x + threadIdx.x;  // HID*HID threads
  const int n = idx >> 8, k = idx & 255;
  Wt[n * HID + k] = (_Float16)W[k * HID + n];
}

// ---------------------------------------------------------------------------
// Kernel 3: h = x + deg_emb[clamp(deg)] + rw_pe @ rw_w + rw_b   (f16 out)
// ---------------------------------------------------------------------------
__global__ void assemble_h_kernel(
    const float* __restrict__ x, const int* __restrict__ degree,
    const float* __restrict__ deg_emb, const float* __restrict__ rw_pe,
    const float* __restrict__ rw_w, const float* __restrict__ rw_b,
    _Float16* __restrict__ h16) {
  const int idx = blockIdx.x * blockDim.x + threadIdx.x;  // NN*HID threads
  const int node = idx >> 8, c = idx & 255;
  int d = degree[node]; d = d > 255 ? 255 : d;
  float v = x[idx] + deg_emb[d * HID + c] + rw_b[c];
  const float* rp = rw_pe + node * K_RW;
#pragma unroll
  for (int k = 0; k < K_RW; ++k) v += rp[k] * rw_w[k * HID + c];
  h16[idx] = (_Float16)v;
}

// ---------------------------------------------------------------------------
// Kernel 4: WMMA GEMM  C[M][Nc] = (A[M][Kd] @ Bt[Nc][Kd]^T + bias) * scale
// Wave computes a 16x64 strip (4 n-tiles).  A fragment prefetched one
// k-step ahead; B fragments software-pipelined inside the k-step.
// ---------------------------------------------------------------------------
template <bool OUT_F16>
__global__ void __launch_bounds__(256) gemm_kernel(
    const _Float16* __restrict__ A, const _Float16* __restrict__ Bt,
    const float* __restrict__ bias, void* __restrict__ Cout,
    int M, int Kd, int Nc, float scale) {
  const int wave = threadIdx.x >> 5, lane = threadIdx.x & 31;
  const int half = lane >> 4, r = lane & 15;
  const int strip = blockIdx.x * 8 + wave;  // one 16x64 strip per wave
  const int nstrips = Nc >> 6;
  const int mt = strip / nstrips, ns = strip - mt * nstrips;
  if (mt * 16 >= M) return;  // uniform per wave

  v8f acc[4] = {};
  const _Float16* arow = A + (size_t)(mt * 16 + r) * Kd;
  const _Float16* brow = Bt + (size_t)(ns * 64 + r) * Kd;

  v16h a = frag_load_contig(arow, half);
  for (int k0 = 0; k0 < Kd; k0 += 32) {
    v16h anext = a;
    if (k0 + 32 < Kd) anext = frag_load_contig(arow + k0 + 32, half);
    v16h bb = frag_load_contig(brow + k0, half);
#pragma unroll
    for (int u = 0; u < 3; ++u) {
      v16h bn = frag_load_contig(brow + (size_t)((u + 1) * 16) * Kd + k0, half);
      acc[u] = wmma_f16(a, bb, acc[u]);
      bb = bn;
    }
    acc[3] = wmma_f16(a, bb, acc[3]);
    a = anext;
  }
#pragma unroll
  for (int u = 0; u < 4; ++u) {
    const int col = ns * 64 + u * 16 + r;
    const float bn = bias[col];
#pragma unroll
    for (int e = 0; e < 8; ++e) {
      const int row = mt * 16 + e + 8 * half;  // D layout: M=e+8*half, N=r
      const float val = (acc[u][e] + bn) * scale;
      const size_t o = (size_t)row * Nc + col;
      if constexpr (OUT_F16) ((_Float16*)Cout)[o] = (_Float16)val;
      else                   ((float*)Cout)[o] = val;
    }
  }
}

// ---------------------------------------------------------------------------
// Kernel 5: fused edge-aware attention, one workgroup per (graph, head).
// scores = (scaled Q) K^T  (single hoisted A fragment, pipelined B)
//        + edge bias scattered via ds_add_f32 LDS atomics (no dense [B,N,N,H])
// split-row softmax (2 threads/row); out = w @ V with V staged TRANSPOSED in
// LDS so every WMMA operand is a contiguous ds_load_b128.  LDS ~132 KB.
// ---------------------------------------------------------------------------
#define QP  40   // padded Q/K row stride (80 B)
#define VTP 136  // padded Vt row stride (272 B)
#define SP  129  // padded f32 score row stride
#define HP  136  // padded f16 softmax row stride

__global__ void __launch_bounds__(256) attention_kernel(
    const _Float16* __restrict__ Q, const _Float16* __restrict__ Km,
    const _Float16* __restrict__ V,
    const int* __restrict__ src, const int* __restrict__ dst,
    const int* __restrict__ gate_type, const float* __restrict__ edge_attr,
    const float* __restrict__ gate_emb, const float* __restrict__ ebp_w,
    const float* __restrict__ ebp_b, _Float16* __restrict__ out) {
  __shared__ __align__(16) _Float16 Qs[N_NODES * QP];
  __shared__ __align__(16) _Float16 Ks[N_NODES * QP];
  __shared__ __align__(16) _Float16 Vt[DH * VTP];      // V transposed: Vt[d][j]
  __shared__ __align__(16) float    S [N_NODES * SP];
  __shared__ __align__(16) _Float16 Sh[N_NODES * HP];
  __shared__ float rmax[256];
  __shared__ float rsum[256];

  const int b = blockIdx.x >> 3;
  const int h = blockIdx.x & 7;
  const int tid = threadIdx.x;

  if (tid == 0) {
    __builtin_prefetch(src + b * E_PER_G, 0, 1);
    __builtin_prefetch(dst + b * E_PER_G, 0, 1);
    __builtin_prefetch(edge_attr + (size_t)b * E_PER_G * 8, 0, 1);
  }

  // stage Q/K head slices row-major (32 halves = 16 dwords per row)
  for (int i = tid; i < N_NODES * 16; i += 256) {
    const int row = i >> 4, dw = i & 15;
    const size_t gofs = (size_t)(b * N_NODES + row) * HID + h * DH;
    ((uint32_t*)Qs)[row * (QP / 2) + dw] = ((const uint32_t*)(Q  + gofs))[dw];
    ((uint32_t*)Ks)[row * (QP / 2) + dw] = ((const uint32_t*)(Km + gofs))[dw];
  }
  // stage V transposed: Vt[d][j] = V[b*128+j][h*32+d] (coalesced global reads)
  for (int i = tid; i < N_NODES * DH; i += 256) {
    const int j = i >> 5, d = i & 31;
    Vt[d * VTP + j] = V[(size_t)(b * N_NODES + j) * HID + h * DH + d];
  }
  __syncthreads();

  const int wave = tid >> 5, lane = tid & 31, half = lane >> 4, r = lane & 15;

  // scores = Q K^T : wave owns a 16x128 strip; Dh=32 -> ONE hoisted A
  // fragment feeds all 8 n-tiles; B pipelined a tile ahead.
  {
    v8f acc[8] = {};
    v16h a = frag_load_contig(Qs + (wave * 16 + r) * QP, half);
    v16h bb = frag_load_contig(Ks + r * QP, half);
#pragma unroll
    for (int t = 0; t < 7; ++t) {
      v16h bn = frag_load_contig(Ks + ((t + 1) * 16 + r) * QP, half);
      acc[t] = wmma_f16(a, bb, acc[t]);
      bb = bn;
    }
    acc[7] = wmma_f16(a, bb, acc[7]);
#pragma unroll
    for (int t = 0; t < 8; ++t)
#pragma unroll
      for (int e = 0; e < 8; ++e)
        S[(wave * 16 + e + 8 * half) * SP + t * 16 + r] = acc[t][e];
  }
  __syncthreads();

  // sparse symmetric edge bias, scattered into LDS score tile (ds_add_f32)
  for (int e = tid; e < E_PER_G; e += 256) {
    const int ge = b * E_PER_G + e;
    const int sl = src[ge] - b * N_NODES;
    const int dl = dst[ge] - b * N_NODES;
    float tb = gate_emb[gate_type[ge] * HEADS + h] + ebp_b[h];
    const float* ea = edge_attr + (size_t)ge * 8;
#pragma unroll
    for (int f = 0; f < 8; ++f) tb += ea[f] * ebp_w[f * HEADS + h];
    atomicAdd(&S[sl * SP + dl], tb);
    if (sl != dl) atomicAdd(&S[dl * SP + sl], tb);
  }
  __syncthreads();

  // split-row softmax: 2 threads per row, 64 columns each
  {
    const int row = tid & 127, part = tid >> 7;
    float* srow = S + row * SP + part * 64;
    float m = srow[0];
    for (int j = 1; j < 64; ++j) m = fmaxf(m, srow[j]);
    rmax[tid] = m;
    __syncthreads();
    const float mrow = fmaxf(rmax[row], rmax[row + 128]);
    float ssum = 0.f;
    for (int j = 0; j < 64; ++j) {
      const float ev = __expf(srow[j] - mrow);
      srow[j] = ev; ssum += ev;
    }
    rsum[tid] = ssum;
    __syncthreads();
    const float inv = 1.f / (rsum[row] + rsum[row + 128]);
    _Float16* hr = Sh + row * HP + part * 64;
    for (int j = 0; j < 64; ++j) hr[j] = (_Float16)(srow[j] * inv);
  }
  __syncthreads();

  // out = w @ V : wave owns a 16x32 strip (2 n-tiles); A prefetched one
  // k-step ahead, B pipelined within the step.
  {
    v8f acc[2] = {};
    const _Float16* arow = Sh + (wave * 16 + r) * HP;
    v16h a = frag_load_contig(arow, half);
#pragma unroll
    for (int k0 = 0; k0 < N_NODES; k0 += 32) {
      v16h anext = a;
      if (k0 + 32 < N_NODES) anext = frag_load_contig(arow + k0 + 32, half);
      v16h bb = frag_load_contig(Vt + r * VTP + k0, half);
      v16h bn = frag_load_contig(Vt + (16 + r) * VTP + k0, half);
      acc[0] = wmma_f16(a, bb, acc[0]);
      acc[1] = wmma_f16(a, bn, acc[1]);
      a = anext;
    }
#pragma unroll
    for (int t = 0; t < 2; ++t)
#pragma unroll
      for (int e = 0; e < 8; ++e)
        out[(size_t)(b * N_NODES + wave * 16 + e + 8 * half) * HID + h * DH +
            t * 16 + r] = (_Float16)acc[t][e];
  }
}

// ---------------------------------------------------------------------------
extern "C" void kernel_launch(void* const* d_in, const int* in_sizes, int n_in,
                              void* d_out, int out_size, void* d_ws, size_t ws_size,
                              hipStream_t stream) {
  (void)in_sizes; (void)n_in; (void)out_size; (void)ws_size;
  const float* x       = (const float*)d_in[0];
  const int*   eidx    = (const int*)d_in[1];
  const int*   esrc    = eidx;
  const int*   edst    = eidx + E_TOT;
  const float* eattr   = (const float*)d_in[2];
  const int*   egate   = (const int*)d_in[3];
  // d_in[4] = batch (derivable, unused)
  const float* demb    = (const float*)d_in[5];
  const float* rww     = (const float*)d_in[6];
  const float* rwb     = (const float*)d_in[7];
  const float* Wq      = (const float*)d_in[8];
  const float* bq      = (const float*)d_in[9];
  const float* Wk      = (const float*)d_in[10];
  const float* bk      = (const float*)d_in[11];
  const float* Wv      = (const float*)d_in[12];
  const float* bv      = (const float*)d_in[13];
  const float* Wo      = (const float*)d_in[14];
  const float* bo      = (const float*)d_in[15];
  const float* gemb    = (const float*)d_in[16];
  const float* ebpw    = (const float*)d_in[17];
  const float* ebpb    = (const float*)d_in[18];

  char* ws = (char*)d_ws;
  size_t off = 0;
  auto carve = [&](size_t bytes) -> void* {
    void* p = ws + off;
    off = (off + bytes + 255) & ~(size_t)255;
    return p;
  };
  int*      degree = (int*)carve((size_t)NN * sizeof(int));
  float*    rwpe   = (float*)carve((size_t)NN * K_RW * sizeof(float));
  _Float16* h16    = (_Float16*)carve((size_t)NN * HID * 2);
  _Float16* Wqt    = (_Float16*)carve((size_t)HID * HID * 2);
  _Float16* Wkt    = (_Float16*)carve((size_t)HID * HID * 2);
  _Float16* Wvt    = (_Float16*)carve((size_t)HID * HID * 2);
  _Float16* Wot    = (_Float16*)carve((size_t)HID * HID * 2);
  _Float16* Qb     = (_Float16*)carve((size_t)NN * HID * 2);
  _Float16* Kb     = (_Float16*)carve((size_t)NN * HID * 2);
  _Float16* Vb     = (_Float16*)carve((size_t)NN * HID * 2);
  _Float16* Ab     = (_Float16*)carve((size_t)NN * HID * 2);
  // total workspace ~82 MB

  rw_pe_kernel<<<B_GRAPHS, 256, 0, stream>>>(esrc, edst, degree, rwpe);

  wt_f16_kernel<<<(HID * HID) / 256, 256, 0, stream>>>(Wq, Wqt);
  wt_f16_kernel<<<(HID * HID) / 256, 256, 0, stream>>>(Wk, Wkt);
  wt_f16_kernel<<<(HID * HID) / 256, 256, 0, stream>>>(Wv, Wvt);
  wt_f16_kernel<<<(HID * HID) / 256, 256, 0, stream>>>(Wo, Wot);

  assemble_h_kernel<<<((size_t)NN * HID) / 256, 256, 0, stream>>>(
      x, degree, demb, rwpe, rww, rwb, h16);

  // one 16x64 strip per wave: (M/16) * (Nc/64) strips, 8 waves per block
  const int gblocks = (NN / 16) * (HID / 64) / 8;
  const float qscale = 0.17677669529663687f;  // 1/sqrt(Dh)
  gemm_kernel<true><<<gblocks, 256, 0, stream>>>(h16, Wqt, bq, Qb, NN, HID, HID, qscale);
  gemm_kernel<true><<<gblocks, 256, 0, stream>>>(h16, Wkt, bk, Kb, NN, HID, HID, 1.f);
  gemm_kernel<true><<<gblocks, 256, 0, stream>>>(h16, Wvt, bv, Vb, NN, HID, HID, 1.f);

  attention_kernel<<<B_GRAPHS * HEADS, 256, 0, stream>>>(
      Qb, Kb, Vb, esrc, edst, egate, eattr, gemb, ebpw, ebpb, Ab);

  gemm_kernel<false><<<gblocks, 256, 0, stream>>>(Ab, Wot, bo, d_out, NN, HID, HID, 1.f);
}
